// GatedDeltaNet_36971078484475
// MI455X (gfx1250) — compile-verified, matrix operand order checked by reading, not run on hardware
//
#include <hip/hip_runtime.h>
#include <hip/hip_bf16.h>
#include <math.h>

// ---------------- problem constants ----------------
#define BB   2
#define TT   2048
#define DD   1024
#define HH   16
#define HD   64
#define KK   4
#define CONV (3 * DD)        // 3072
#define MM   (BB * TT)       // 4096 rows

typedef __bf16 bf16_t;
typedef __attribute__((ext_vector_type(8)))  __bf16 v8bf;
typedef __attribute__((ext_vector_type(16))) __bf16 v16bf;
typedef __attribute__((ext_vector_type(8)))  float  v8f;
typedef __attribute__((ext_vector_type(4)))  unsigned int v4u;
typedef __attribute__((ext_vector_type(8)))  int v8i;
typedef __attribute__((ext_vector_type(4)))  int v4i;

// ---------------- fp32 -> bf16 convert ----------------
__global__ void k_cvt_bf16(const float* __restrict__ s, bf16_t* __restrict__ d, int n) {
    int i = blockIdx.x * blockDim.x + threadIdx.x;
    if (i < n) d[i] = (bf16_t)s[i];
}

// ---------------- TDM 2D tile load: global (rows x cols bf16, row stride) -> LDS ----------------
// D# per CDNA5 ISA:
//   group0: [1:0]=count=1, [63:32]=lds_addr, [120:64]=global_addr, [127:126]=type=2
//   group1: [17:16]=data_size(1=2B), [79:48]=tensor_dim0, [111:80]=tensor_dim1,
//           [127:112]=tile_dim0, [143:128]=tile_dim1, [207:160]=tensor_dim0_stride
// This toolchain exposes the 6-arg builtin: (v4u g0, v8i g1, v4i g2, v4i g3, v8i, i32 cpol)
__device__ __forceinline__ void tdm_load_tile(unsigned lds_off, const bf16_t* gptr,
                                              int rows, int cols, int row_stride) {
    unsigned long long ga = (unsigned long long)(uintptr_t)gptr;
    v4u g0 = { 1u,                                   // count=1, user descriptor
               lds_off,                              // LDS byte address
               (unsigned)(ga & 0xffffffffu),         // global_addr[31:0]
               (unsigned)(((ga >> 32) & 0x01ffffffu) // global_addr[56:32]
                          | 0x80000000u) };          // type=2 ("image")
    v8i g1;
    g1[0] = (1 << 16);                    // data_size = 1 (2 bytes); mask/flags = 0
    g1[1] = (cols & 0xffff) << 16;        // tensor_dim0[15:0]
    g1[2] = (rows & 0xffff) << 16;        // tensor_dim0[31:16]=0 | tensor_dim1[15:0]
    g1[3] = (cols & 0xffff) << 16;        // tensor_dim1[31:16]=0 | tile_dim0
    g1[4] = (rows & 0xffff);              // tile_dim1 | tile_dim2=0
    g1[5] = row_stride;                   // tensor_dim0_stride[31:0] (elements)
    g1[6] = 0;                            // stride hi | tensor_dim1_stride lo
    g1[7] = 0;
    v4i z4 = { 0, 0, 0, 0 };              // groups 2/3 unused (2D tensor)
    v8i z8 = { 0, 0, 0, 0, 0, 0, 0, 0 };
    __builtin_amdgcn_tensor_load_to_lds(g0, g1, z4, z4, z8, 0);
}

// ---------------- WMMA bf16 GEMM:  C[M,N] = A[M,K] * W[N,K]^T ----------------
// Block = 256 thr (8 waves, 2x4), block tile 128x128, wave tile 64x32.
// Per k-step TDM DMAs A-tile(128x32) and B-tile(128x32) into LDS; fragments read via ds_load_b128.
// 16-bit A fragment layout (ISA 7.12.2): lane L (L<16): row M=L, elems 0..7 = K+0..7,
// elems 8..15 = K+16..23 ; lane L+16: row M=L, K+8..15 and K+24..31. B loads identically from W rows.
__device__ __forceinline__ v16bf lds_frag(const bf16_t* __restrict__ base, int lane) {
    const bf16_t* p = base + (lane & 15) * 32 + ((lane >> 4) << 3);
    v8bf lo = *(const v8bf*)(p);
    v8bf hi = *(const v8bf*)(p + 16);
    return __builtin_shufflevector(lo, hi, 0,1,2,3,4,5,6,7,8,9,10,11,12,13,14,15);
}

__global__ __launch_bounds__(256) void k_gemm_bf16(const bf16_t* __restrict__ A,
                                                   const bf16_t* __restrict__ W,
                                                   float* __restrict__ C,
                                                   int M, int N, int Kd) {
    __shared__ bf16_t Atile[128 * 32];
    __shared__ bf16_t Btile[128 * 32];

    const int lane = threadIdx.x & 31;
    const int wave = threadIdx.x >> 5;
    const int wm = wave >> 2;               // 0..1
    const int wn = wave & 3;                // 0..3
    const int mblk = blockIdx.y * 128;
    const int nblk = blockIdx.x * 128;

    const unsigned aoff = (unsigned)(uintptr_t)(&Atile[0]);
    const unsigned boff = (unsigned)(uintptr_t)(&Btile[0]);

    v8f acc[4][2];
#pragma unroll
    for (int tm = 0; tm < 4; ++tm)
#pragma unroll
        for (int tn = 0; tn < 2; ++tn) acc[tm][tn] = (v8f){};

    for (int k0 = 0; k0 < Kd; k0 += 32) {
        __syncthreads();                    // previous iteration's LDS reads done
        if (wave == 0) {
            tdm_load_tile(aoff, A + (size_t)mblk * Kd + k0, 128, 32, Kd);
            tdm_load_tile(boff, W + (size_t)nblk * Kd + k0, 128, 32, Kd);
            __builtin_amdgcn_s_wait_tensorcnt(0);
        }
        __syncthreads();                    // tiles visible to all waves

        v16bf af[4], bf[2];
#pragma unroll
        for (int tm = 0; tm < 4; ++tm)
            af[tm] = lds_frag(&Atile[(wm * 64 + tm * 16) * 32], lane);
#pragma unroll
        for (int tn = 0; tn < 2; ++tn)
            bf[tn] = lds_frag(&Btile[(wn * 32 + tn * 16) * 32], lane);

#pragma unroll
        for (int tm = 0; tm < 4; ++tm)
#pragma unroll
            for (int tn = 0; tn < 2; ++tn)
                acc[tm][tn] = __builtin_amdgcn_wmma_f32_16x16x32_bf16(
                    false, af[tm], false, bf[tn], (short)0, acc[tm][tn], false, false);
    }

    // D layout: lane = ((m>>3)<<4) | n, vgpr r = m&7
    const int cn = lane & 15;
    const int mh = (lane >> 4) * 8;
#pragma unroll
    for (int tm = 0; tm < 4; ++tm)
#pragma unroll
        for (int tn = 0; tn < 2; ++tn)
#pragma unroll
            for (int r = 0; r < 8; ++r)
                C[(size_t)(mblk + wm * 64 + tm * 16 + mh + r) * N +
                  (nblk + wn * 32 + tn * 16 + cn)] = acc[tm][tn][r];
}

// ---------------- beta / g projections (wave per (bt,h)) ----------------
__device__ __forceinline__ float wave_sum(float v) {
#pragma unroll
    for (int off = 16; off > 0; off >>= 1) v += __shfl_xor(v, off, 32);
    return v;
}

__global__ __launch_bounds__(256) void k_small_proj(const float* __restrict__ x,
                                                    const float* __restrict__ Wb,
                                                    const float* __restrict__ Wa,
                                                    const float* __restrict__ dtb,
                                                    const float* __restrict__ Alog,
                                                    float* __restrict__ beta,
                                                    float* __restrict__ g) {
    const int lane = threadIdx.x & 31;
    const int gw = blockIdx.x * 8 + (threadIdx.x >> 5);   // global wave
    const int bt = gw >> 4;
    const int h  = gw & 15;
    const float* xr = x + (size_t)bt * DD;
    const float* wb = Wb + (size_t)h * DD;
    const float* wa = Wa + (size_t)h * DD;
    float s1 = 0.f, s2 = 0.f;
    for (int i = lane; i < DD; i += 32) {
        float xv = xr[i];
        s1 += xv * wb[i];
        s2 += xv * wa[i];
    }
    s1 = wave_sum(s1);
    s2 = wave_sum(s2);
    if (lane == 0) {
        beta[bt * HH + h] = 1.f / (1.f + __expf(-s1));
        float a = s2 + dtb[h];
        float sp = (a > 20.f) ? a : log1pf(__expf(a));
        g[bt * HH + h] = -__expf(Alog[h]) * sp;
    }
}

// ---------------- causal depthwise conv (K=4) + SiLU ----------------
__global__ void k_conv_silu(const float* __restrict__ mixed,
                            const float* __restrict__ cw,
                            float* __restrict__ out) {
    int idx = blockIdx.x * blockDim.x + threadIdx.x;
    if (idx >= MM * CONV) return;
    int c  = idx % CONV;
    int bt = idx / CONV;
    int t  = bt % TT;
    float acc = 0.f;
#pragma unroll
    for (int j = 0; j < KK; ++j) {
        int ts = t + j - (KK - 1);
        if (ts >= 0) acc += mixed[(size_t)(bt + j - (KK - 1)) * CONV + c] * cw[c * KK + j];
    }
    out[idx] = acc / (1.f + __expf(-acc));
}

// ---------------- l2norm of q,k heads (wave per (bt,h)) ----------------
__global__ __launch_bounds__(256) void k_l2norm(const float* __restrict__ conv,
                                                float* __restrict__ qn,
                                                float* __restrict__ kn) {
    const int lane = threadIdx.x & 31;
    const int gw = blockIdx.x * 8 + (threadIdx.x >> 5);
    const int bt = gw >> 4;
    const int h  = gw & 15;
    const float* row = conv + (size_t)bt * CONV;
    float q0 = row[h * HD + lane], q1 = row[h * HD + lane + 32];
    float k0 = row[DD + h * HD + lane], k1 = row[DD + h * HD + lane + 32];
    float sq = wave_sum(q0 * q0 + q1 * q1);
    float sk = wave_sum(k0 * k0 + k1 * k1);
    float iq = rsqrtf(sq + 1e-6f) * 0.125f;   // * HD^-0.5
    float ik = rsqrtf(sk + 1e-6f);
    size_t o = (size_t)bt * DD + h * HD;
    qn[o + lane] = q0 * iq;  qn[o + lane + 32] = q1 * iq;
    kn[o + lane] = k0 * ik;  kn[o + lane + 32] = k1 * ik;
}

// ---------------- sequential gated delta-rule scan ----------------
// one block per (b,h); state S[64][64] fp32 in registers (16 per thread)
__global__ __launch_bounds__(256) void k_scan(const float* __restrict__ qn,
                                              const float* __restrict__ kn,
                                              const float* __restrict__ conv,
                                              const float* __restrict__ beta,
                                              const float* __restrict__ g,
                                              float* __restrict__ attn) {
    const int b = blockIdx.x / HH;
    const int h = blockIdx.x % HH;
    const int tid = threadIdx.x;
    const int vc = tid & 63;          // v column
    const int qd = tid >> 6;          // d quarter 0..3
    const int d0 = qd * 16;

    __shared__ float sq[HD], sk[HD], sv[HD], sgb[2];
    __shared__ float sred[4][HD];

    float S[16];
#pragma unroll
    for (int i = 0; i < 16; ++i) S[i] = 0.f;

    for (int t = 0; t < TT; ++t) {
        const size_t row = (size_t)b * TT + t;
        __syncthreads();
        if (tid < 64)        sq[tid]       = qn[row * DD + h * HD + tid];
        else if (tid < 128)  sk[tid - 64]  = kn[row * DD + h * HD + (tid - 64)];
        else if (tid < 192)  sv[tid - 128] = conv[row * CONV + 2 * DD + h * HD + (tid - 128)];
        else if (tid == 192) sgb[0] = g[row * HH + h];
        else if (tid == 193) sgb[1] = beta[row * HH + h];
        __syncthreads();

        float eg = __expf(sgb[0]);
        float vb = sv[vc] * sgb[1];
        float po = 0.f;
#pragma unroll
        for (int i = 0; i < 16; ++i) {
            S[i] = S[i] * eg + sk[d0 + i] * vb;
            po += sq[d0 + i] * S[i];
        }
        sred[qd][vc] = po;
        __syncthreads();
        if (qd == 0)
            attn[row * DD + h * HD + vc] =
                sred[0][vc] + sred[1][vc] + sred[2][vc] + sred[3][vc];
    }
}

// ---------------- gated RMSNorm, gate = silu(z); emit bf16 ----------------
__global__ __launch_bounds__(256) void k_gated_norm(const float* __restrict__ attn,
                                                    const float* __restrict__ z,
                                                    bf16_t* __restrict__ outb) {
    const int lane = threadIdx.x & 31;
    const int gw = blockIdx.x * 8 + (threadIdx.x >> 5);
    const int bt = gw >> 4;
    const int h  = gw & 15;
    size_t o = (size_t)bt * DD + h * HD;
    float a0 = attn[o + lane], a1 = attn[o + lane + 32];
    float ss = wave_sum(a0 * a0 + a1 * a1);
    float inv = rsqrtf(ss * (1.f / HD) + 1e-6f);
    float z0 = z[o + lane], z1 = z[o + lane + 32];
    float g0 = z0 / (1.f + __expf(-z0));
    float g1 = z1 / (1.f + __expf(-z1));
    outb[o + lane]      = (bf16_t)(a0 * inv * g0);
    outb[o + lane + 32] = (bf16_t)(a1 * inv * g1);
}

// ---------------- host side ----------------
extern "C" void kernel_launch(void* const* d_in, const int* in_sizes, int n_in,
                              void* d_out, int out_size, void* d_ws, size_t ws_size,
                              hipStream_t stream) {
    const float* x    = (const float*)d_in[0];
    const float* Wqkv = (const float*)d_in[1];
    const float* cw   = (const float*)d_in[2];
    const float* Wz   = (const float*)d_in[3];
    const float* Wb   = (const float*)d_in[4];
    const float* Wa   = (const float*)d_in[5];
    const float* dtb  = (const float*)d_in[6];
    const float* Alog = (const float*)d_in[7];
    const float* Wout = (const float*)d_in[8];
    float* out = (float*)d_out;

    char* ws = (char*)d_ws;
    size_t off = 0;
    auto alloc = [&](size_t bytes) -> char* {
        char* p = ws + off;
        off = (off + bytes + 255) & ~(size_t)255;
        return p;
    };

    bf16_t* xb    = (bf16_t*)alloc((size_t)MM * DD * 2);
    bf16_t* wqkvb = (bf16_t*)alloc((size_t)CONV * DD * 2);
    bf16_t* wzb   = (bf16_t*)alloc((size_t)DD * DD * 2);
    bf16_t* woutb = (bf16_t*)alloc((size_t)DD * DD * 2);
    float*  mixed = (float*)alloc((size_t)MM * CONV * 4);
    float*  convb = (float*)alloc((size_t)MM * CONV * 4);
    float*  zbuf  = (float*)alloc((size_t)MM * DD * 4);
    float*  qn    = (float*)alloc((size_t)MM * DD * 4);
    float*  kn    = (float*)alloc((size_t)MM * DD * 4);
    float*  beta  = (float*)alloc((size_t)MM * HH * 4);
    float*  gbuf  = (float*)alloc((size_t)MM * HH * 4);
    float*  attn  = (float*)alloc((size_t)MM * DD * 4);
    bf16_t* normb = (bf16_t*)alloc((size_t)MM * DD * 2);

    // fp32 -> bf16 converts
    k_cvt_bf16<<<(MM * DD + 255) / 256, 256, 0, stream>>>(x, xb, MM * DD);
    k_cvt_bf16<<<(CONV * DD + 255) / 256, 256, 0, stream>>>(Wqkv, wqkvb, CONV * DD);
    k_cvt_bf16<<<(DD * DD + 255) / 256, 256, 0, stream>>>(Wz, wzb, DD * DD);
    k_cvt_bf16<<<(DD * DD + 255) / 256, 256, 0, stream>>>(Wout, woutb, DD * DD);

    // qkv projection + z projection (WMMA + TDM)
    dim3 gq(CONV / 128, MM / 128);
    k_gemm_bf16<<<gq, 256, 0, stream>>>(xb, wqkvb, mixed, MM, CONV, DD);
    dim3 gz(DD / 128, MM / 128);
    k_gemm_bf16<<<gz, 256, 0, stream>>>(xb, wzb, zbuf, MM, DD, DD);

    // beta / g
    k_small_proj<<<MM * HH / 8, 256, 0, stream>>>(x, Wb, Wa, dtb, Alog, beta, gbuf);

    // causal depthwise conv + silu
    k_conv_silu<<<(MM * CONV + 255) / 256, 256, 0, stream>>>(mixed, cw, convb);

    // l2norm q,k
    k_l2norm<<<MM * HH / 8, 256, 0, stream>>>(convb, qn, kn);

    // gated delta-rule scan
    k_scan<<<BB * HH, 256, 0, stream>>>(qn, kn, convb, beta, gbuf, attn);

    // gated RMSNorm -> bf16
    k_gated_norm<<<MM * HH / 8, 256, 0, stream>>>(attn, zbuf, normb);

    // output projection (WMMA + TDM) -> d_out fp32
    dim3 go(DD / 128, MM / 128);
    k_gemm_bf16<<<go, 256, 0, stream>>>(normb, woutb, out, MM, DD, DD);
}
